// Decoder_Atten_80032420593886
// MI455X (gfx1250) — matrix operand order repaired and checked by reading, not compiled, not run on hardware
//
#include <hip/hip_runtime.h>
#include <hip/hip_bf16.h>

#define B_   32
#define H_   512
#define E_   512
#define V_   32000
#define L_   512
#define T_   64
#define KC_  1536   // concat [emb(512) | ctx(512) | h(512)]
#define G4_  2048   // 4*H

typedef __attribute__((ext_vector_type(2)))  float        v2f;
typedef __attribute__((ext_vector_type(8)))  float        v8f;
typedef __attribute__((ext_vector_type(16))) __bf16       v16bf;
typedef __attribute__((ext_vector_type(4)))  unsigned int v4u;
typedef __attribute__((ext_vector_type(8)))  int          v8i;
typedef __attribute__((ext_vector_type(4)))  int          v4i;

#define LDS_ROW_PITCH 144   // 128B tile row + 16B TDM pad -> conflict-free ds_load_b128
#define LDS_TILE_BYTES (128 * LDS_ROW_PITCH)   // 18432 B per buffer

__device__ __forceinline__ unsigned short f32_to_bf16(float f) {
  unsigned int u = __float_as_uint(f);
  u += 0x7fffu + ((u >> 16) & 1u);      // round-to-nearest-even
  return (unsigned short)(u >> 16);
}
__device__ __forceinline__ float sigmoidf_(float x) { return 1.0f / (1.0f + __expf(-x)); }
__device__ __forceinline__ unsigned lds_off_of(const void* p) {
  // generic pointer to LDS: low 32 bits are the LDS byte offset
  return (unsigned)(unsigned long long)p;
}

// TDM: load a 2D tile (128 rows x 128 B) of Hs into LDS,
// inserting 16B of padding after every 128B row (LDS pitch 144B).
__device__ __forceinline__ void tdm_load_tile(const void* gptr, unsigned lds_addr) {
  unsigned long long ga = (unsigned long long)gptr;
  v4u g0;
  g0[0] = 1u;                                  // count=1 (valid), no gather
  g0[1] = lds_addr;                            // LDS byte address
  g0[2] = (unsigned)ga;                        // global_addr[31:0]
  g0[3] = (unsigned)((ga >> 32) & 0x01FFFFFFu) // global_addr[56:32]
          | 0x80000000u;                       // type=2 ("image") in bits[127:126]
  v8i g1;
  g1[0] = (int)((3u << 16)                     // data_size=3 (8B units)
          | (1u << 20)                         // pad_enable
          | (4u << 22)                         // pad_interval: 32 DWORDs (=128B row)
          | (3u << 25));                       // pad_amount: 4 DWORDs (=16B)
  g1[1] = (int)(16u << 16);                    // tensor_dim0 = 16 units (128B row)
  g1[2] = (int)(128u << 16);                   // tensor_dim1 = 128 rows
  g1[3] = (int)(16u << 16);                    // tile_dim0 = 16 units (128B)
  g1[4] = (int)128u;                           // tile_dim1 = 128 rows, tile_dim2 = 0
  g1[5] = (int)128u;                           // tensor_dim0_stride = 128 units (1024B = Hs row)
  g1[6] = 0;
  g1[7] = 0;
  v4i z4 = {0, 0, 0, 0};                       // groups 2/3 unused (2D tile)
  v8i z8 = {0, 0, 0, 0, 0, 0, 0, 0};
  __builtin_amdgcn_tensor_load_to_lds(g0, g1, z4, z4, z8, 0);
}

// ---------------- init: copy h0/c0 into state buffers ----------------
__global__ void k_init(const float* __restrict__ h0, const float* __restrict__ c0,
                       float* __restrict__ hbuf, float* __restrict__ cbuf) {
  int i = blockIdx.x * 256 + threadIdx.x;
  if (i < B_ * H_) { hbuf[i] = h0[i]; cbuf[i] = c0[i]; }
}

// ---------------- one-time W_out f32 -> bf16 ----------------
__global__ void k_w2bf(const float* __restrict__ W, unsigned short* __restrict__ Wb, int n) {
  for (int i = blockIdx.x * 256 + threadIdx.x; i < n; i += gridDim.x * 256)
    Wb[i] = f32_to_bf16(W[i]);
}

// ---------------- per-step attention + build z = [emb|ctx|h] ----------------
__global__ void k_attn(const float* __restrict__ enc, const int* __restrict__ mask_att,
                       const int* __restrict__ x, const int* __restrict__ targets,
                       const float* __restrict__ emb_tbl, const float* __restrict__ hbuf,
                       float* __restrict__ z, int t) {
  const int b = blockIdx.x, tid = threadIdx.x;
  __shared__ float s_h[H_];
  __shared__ float s_w[L_];
  __shared__ float red[256];
  for (int i = tid; i < H_; i += 256) s_h[i] = hbuf[b * H_ + i];
  __syncthreads();
  const float* encb = enc + (size_t)b * L_ * H_;
  const float NEG = -__builtin_inff();
  for (int l = tid; l < L_; l += 256) {
    const float4* row = (const float4*)(encb + (size_t)l * H_);
    const float4* hv  = (const float4*)s_h;
    float acc = 0.f;
#pragma unroll 4
    for (int k = 0; k < H_ / 4; ++k) {
      float4 r = row[k], hh = hv[k];
      acc += r.x * hh.x + r.y * hh.y + r.z * hh.z + r.w * hh.w;
    }
    s_w[l] = (mask_att[b * L_ + l] == 0) ? NEG : acc;
  }
  __syncthreads();
  red[tid] = fmaxf(s_w[tid], s_w[tid + 256]);
  __syncthreads();
  for (int s = 128; s > 0; s >>= 1) { if (tid < s) red[tid] = fmaxf(red[tid], red[tid + s]); __syncthreads(); }
  float mx = red[0]; __syncthreads();
  float e0 = __expf(s_w[tid] - mx), e1 = __expf(s_w[tid + 256] - mx);
  s_w[tid] = e0; s_w[tid + 256] = e1;
  red[tid] = e0 + e1;
  __syncthreads();
  for (int s = 128; s > 0; s >>= 1) { if (tid < s) red[tid] += red[tid + s]; __syncthreads(); }
  float inv = 1.0f / red[0]; __syncthreads();
  s_w[tid] *= inv; s_w[tid + 256] *= inv;
  __syncthreads();
  const int tok = (t == 0) ? x[b] : targets[b * T_ + (t - 1)];
  const int i0 = tid, i1 = tid + 256;
  z[b * KC_ + i0] = emb_tbl[(size_t)tok * E_ + i0];
  z[b * KC_ + i1] = emb_tbl[(size_t)tok * E_ + i1];
  float a0 = 0.f, a1 = 0.f;
  for (int l = 0; l < L_; ++l) {
    float w = s_w[l];
    const float* row = encb + (size_t)l * H_;
    a0 += w * row[i0];
    a1 += w * row[i1];
  }
  z[b * KC_ + E_ + i0] = a0;
  z[b * KC_ + E_ + i1] = a1;
  z[b * KC_ + E_ + H_ + i0] = s_h[i0];
  z[b * KC_ + E_ + H_ + i1] = s_h[i1];
}

// ---------------- per-step gates GEMM: (32 x 2048 x 1536) via f32 WMMA 16x16x4 ----------------
__global__ void k_gates(const float* __restrict__ z, const float* __restrict__ W_ih,
                        const float* __restrict__ W_hh, const float* __restrict__ b_ih,
                        const float* __restrict__ b_hh, float* __restrict__ gates) {
  const int wave = blockIdx.x * 8 + (threadIdx.x >> 5);   // 256 waves = 2 Mtiles x 128 Ntiles
  const int lane = threadIdx.x & 31;
  const int mt = wave & 1, nt = wave >> 1;
  const int m0 = mt * 16, j0 = nt * 16;
  const int r = lane & 15, half = lane >> 4;
  const int koff = half * 2;
  v8f acc = {};
  const float* zr  = z    + (size_t)(m0 + r) * KC_;
  const float* wih = W_ih + (size_t)(j0 + r) * (E_ + H_);
  const float* whh = W_hh + (size_t)(j0 + r) * H_;
  for (int k0 = 0; k0 < E_ + H_; k0 += 4) {
    v2f a  = *(const v2f*)(zr  + k0 + koff);
    v2f bb = *(const v2f*)(wih + k0 + koff);
    acc = __builtin_amdgcn_wmma_f32_16x16x4_f32(false, a, false, bb, (short)0, acc, false, false);
  }
  for (int k0 = 0; k0 < H_; k0 += 4) {
    v2f a  = *(const v2f*)(zr  + (E_ + H_) + k0 + koff);
    v2f bb = *(const v2f*)(whh + k0 + koff);
    acc = __builtin_amdgcn_wmma_f32_16x16x4_f32(false, a, false, bb, (short)0, acc, false, false);
  }
  const int j = j0 + r;
  const float bias = b_ih[j] + b_hh[j];
#pragma unroll
  for (int rr = 0; rr < 8; ++rr) {
    int m = m0 + half * 8 + rr;
    gates[m * G4_ + j] = acc[rr] + bias;
  }
}

// ---------------- per-step LSTM cell + bf16 hidden emit ----------------
// Hs stored in plain K order: within each 32-wide K block,
// lane half h's 16-element A frag is the contiguous K range 16h..16h+15.
__global__ void k_cell(const float* __restrict__ gates, float* __restrict__ cbuf,
                       float* __restrict__ hbuf, unsigned short* __restrict__ Hs, int t) {
  int idx = blockIdx.x * 256 + threadIdx.x;   // B*H = 16384
  int b = idx >> 9, hh = idx & 511;
  const float* g = gates + b * G4_;
  float gi = g[hh], gf = g[H_ + hh], gg = g[2 * H_ + hh], go = g[3 * H_ + hh];
  float cn = sigmoidf_(gf) * cbuf[idx] + sigmoidf_(gi) * tanhf(gg);
  float hn = sigmoidf_(go) * tanhf(cn);
  cbuf[idx] = cn;
  hbuf[idx] = hn;
  int ko = hh & 31;
  int pos = hh;
  Hs[(size_t)(t * B_ + b) * H_ + pos] = f32_to_bf16(hn);
}

// ---------------- final logits GEMM: (2048 x 32000 x 512) bf16 WMMA, TDM-staged A ----------------
// Block = 1 M-block (128 rows, shared via LDS) x 8 N-tiles (1 per wave).
__global__ void k_logits(const __bf16* __restrict__ Hs, const __bf16* __restrict__ Wb,
                         const float* __restrict__ b_out, const unsigned char* __restrict__ mask_logit,
                         float* __restrict__ out) {
  __shared__ __attribute__((aligned(128))) unsigned char tile[2][LDS_TILE_BYTES];
  const int warp = threadIdx.x >> 5;
  const int lane = threadIdx.x & 31;
  const int mb = blockIdx.x / 250;                  // 0..15
  const int nt = (blockIdx.x % 250) * 8 + warp;     // 0..1999
  const int m_base = mb * 128;
  const int v0 = nt * 16;
  const int r = lane & 15, half = lane >> 4;
  const int v = v0 + r;

  // prologue: wave 0 DMAs chunk 0 (128 rows x 64 K = 128B/row) into LDS buffer 0
  if (warp == 0) {
    tdm_load_tile(Hs + (size_t)m_base * H_, lds_off_of(&tile[0][0]));
    __builtin_amdgcn_s_wait_tensorcnt(0);
  }
  __syncthreads();

  v8f acc[8] = {};
  const __bf16* wrow = Wb + (size_t)v * H_;
  for (int c = 0; c < 8; ++c) {                     // 8 chunks of 64 K
    const unsigned char* lbase = &tile[c & 1][0];
    if (warp == 0 && c + 1 < 8)                     // prefetch next chunk via TDM
      tdm_load_tile(Hs + (size_t)m_base * H_ + (c + 1) * 64, lds_off_of(&tile[(c + 1) & 1][0]));
    const int k0 = c * 64;
#pragma unroll
    for (int j = 0; j < 2; ++j) {                   // two 32-wide K blocks per chunk
      v16bf bfrag = *(const v16bf*)(wrow + k0 + j * 32 + half * 16);
      if (c + 1 < 8) __builtin_prefetch(wrow + k0 + 64 + j * 32 + half * 16, 0, 1);
#pragma unroll
      for (int i = 0; i < 8; ++i) {                 // 8 M tiles
        const v16bf afrag =
            *(const v16bf*)(lbase + (i * 16 + r) * LDS_ROW_PITCH + j * 64 + half * 32);
        acc[i] = __builtin_amdgcn_wmma_f32_16x16x32_bf16(false, afrag, false, bfrag,
                                                         (short)0, acc[i], false, false);
      }
    }
    if (warp == 0 && c + 1 < 8) __builtin_amdgcn_s_wait_tensorcnt(0);
    __syncthreads();
  }

  const float bias = b_out[v];
  const bool masked = (mask_logit[v] != 0);
  const float NEG = -__builtin_inff();
#pragma unroll
  for (int i = 0; i < 8; ++i) {
#pragma unroll
    for (int rr = 0; rr < 8; ++rr) {
      int m = m_base + i * 16 + half * 8 + rr;      // m = t*32 + b
      int tt = m >> 5, bb = m & 31;
      out[((size_t)bb * T_ + tt) * V_ + v] = masked ? NEG : (acc[i][rr] + bias);
    }
  }
}

extern "C" void kernel_launch(void* const* d_in, const int* in_sizes, int n_in,
                              void* d_out, int out_size, void* d_ws, size_t ws_size,
                              hipStream_t stream) {
  const int*   x        = (const int*)  d_in[0];
  const float* enc      = (const float*)d_in[1];
  const float* h0       = (const float*)d_in[2];
  const float* c0       = (const float*)d_in[3];
  const int*   targets  = (const int*)  d_in[4];
  const int*   mask_att = (const int*)  d_in[5];
  // d_in[6] = teacher_forcing_ratio (==1, teacher forcing always on)
  const float* emb      = (const float*)d_in[7];
  const float* W_ih     = (const float*)d_in[8];
  const float* W_hh     = (const float*)d_in[9];
  const float* b_ih     = (const float*)d_in[10];
  const float* b_hh     = (const float*)d_in[11];
  const float* W_out    = (const float*)d_in[12];
  const float* b_out    = (const float*)d_in[13];
  const unsigned char* mask_logit = (const unsigned char*)d_in[14];
  float* out = (float*)d_out;

  char* ws = (char*)d_ws;
  float* hbuf  = (float*)ws;                 ws += (size_t)B_ * H_ * 4;      // 64 KB
  float* cbuf  = (float*)ws;                 ws += (size_t)B_ * H_ * 4;      // 64 KB
  float* zbuf  = (float*)ws;                 ws += (size_t)B_ * KC_ * 4;     // 192 KB
  float* gates = (float*)ws;                 ws += (size_t)B_ * G4_ * 4;     // 256 KB
  unsigned short* Hs = (unsigned short*)ws;  ws += (size_t)T_ * B_ * H_ * 2; // 2 MB
  unsigned short* Wb = (unsigned short*)ws;                                   // 32.75 MB

  k_init<<<64, 256, 0, stream>>>(h0, c0, hbuf, cbuf);
  k_w2bf<<<4096, 256, 0, stream>>>(W_out, Wb, V_ * H_);

  for (int t = 0; t < T_; ++t) {
    k_attn <<<B_, 256, 0, stream>>>(enc, mask_att, x, targets, emb, hbuf, zbuf, t);
    k_gates<<<32, 256, 0, stream>>>(zbuf, W_ih, W_hh, b_ih, b_hh, gates);
    k_cell <<<64, 256, 0, stream>>>(gates, cbuf, hbuf, Hs, t);
  }

  k_logits<<<4000, 256, 0, stream>>>((const __bf16*)Hs, (const __bf16*)Wb,
                                     b_out, mask_logit, out);
}
